// Backflow_70471823393186
// MI455X (gfx1250) — matrix operand order repaired and checked by reading, not compiled, not run on hardware
//
#include <hip/hip_runtime.h>

typedef __attribute__((ext_vector_type(16))) _Float16 v16h;
typedef __attribute__((ext_vector_type(8)))  float    v8f;

#define NPART   15
#define NPAIR   210          // 15*14 ordered pairs, diagonal excluded
#define NUITER  7            // 7 iterations x 32 pairs = 224 (padded)
#define DBAS    64
#define CUTOFFF 10.0f
#define LN2F     0.69314718055994530942f
#define LOG2EF   1.44269504088896340736f
#define SQLOG2EF 1.20112240878644980f   // sqrt(log2 e)

// shifted softplus via raw v_exp_f32 / v_log_f32 (log2-domain):
// ln(e^x + 1) + ln(1/2) = ln2 * (log2(2^(x*log2e) + 1) - 1)
__device__ __forceinline__ float ssp(float x) {
    float e = __builtin_amdgcn_exp2f(x * LOG2EF);
    float l = __builtin_amdgcn_logf(e + 1.0f);
    return __builtin_fmaf(l, LN2F, -LN2F);
}

// compile-time basis constants (per K index)
__host__ __device__ constexpr float Qk(int K)   { return (float)(2 * K + 1) * (1.0f / 128.0f); }
__host__ __device__ constexpr float ISGk(int K) { return 7.0f * SQLOG2EF / (1.0f + 10.0f * Qk(K)); }
__host__ __device__ constexpr float NMUk(int K) { return -10.0f * Qk(K) * Qk(K) * ISGk(K); }

__global__ __launch_bounds__(32) void Backflow_70471823393186_kernel(
    const float* __restrict__ rs,   // (B, 30, 3)
    const float* __restrict__ W0,   // (3, 64, 16)
    const float* __restrict__ b0,   // (3, 16)
    const float* __restrict__ W1,   // (3, 16, 4)
    const float* __restrict__ b1,   // (3, 4)
    const float* __restrict__ W2,   // (3, 4, 1)
    float* __restrict__ out)        // (B, 30, 3)
{
    __shared__ float xs[48];            // 15 particles x 3
    __shared__ float acc[48];           // backflow accumulator
    __shared__ float h1buf[32 * 17];    // two 16x16 h1 tiles, stride 17

    const int l   = threadIdx.x;        // 0..31 (wave32)
    const int blk = blockIdx.x;         // 0..2*B-1
    const float* base = rs + (size_t)blk * (NPART * 3);

    for (int t = l; t < NPART * 3; t += 32) xs[t] = base[t];
    __syncthreads();

    const int lh = l & 15;              // lane within half-wave
    const int hi = (l >> 4) & 1;        // 0 = lanes 0-15, 1 = lanes 16-31

    // per-lane basis constants for the 32 (chunk,h) A-fragment slots:
    // slot sl = 16*c + h holds K = 32c + (h&7) + ((h&8)?16:0) + 8*hi
    // t = fma(d, isg, nmu);  basis = env * exp2(-(t*t))
    float isg[32], nmu[32];
    #pragma unroll
    for (int c = 0; c < 2; ++c) {
        #pragma unroll
        for (int h = 0; h < 16; ++h) {
            const int K0 = 32 * c + (h & 7) + ((h & 8) ? 16 : 0);
            const int sl = 16 * c + h;
            isg[sl] = hi ? ISGk(K0 + 8) : ISGk(K0);
            nmu[sl] = hi ? NMUk(K0 + 8) : NMUk(K0);
        }
    }

    for (int k = 0; k < 3; ++k) {
        // ---- W0[k] -> two B-fragments (32x16 f16, documented K/N lane layout) ----
        const float* w0p = W0 + k * DBAS * 16;
        v16h bf0, bf1;
        #pragma unroll
        for (int h = 0; h < 16; ++h) {
            int K = (h & 7) + ((h & 8) ? 16 : 0) + (hi ? 8 : 0);
            bf0[h] = (_Float16)w0p[(K)      * 16 + lh];
            bf1[h] = (_Float16)w0p[(K + 32) * 16 + lh];
        }
        const float b0v  = b0[k * 16 + lh];      // D-layout: lane holds column N = lh
        const float* w1p = W1 + k * 16 * 4;
        const float* b1p = b1 + k * 4;
        const float* w2p = W2 + k * 4;

        for (int t = l; t < NPART * 3; t += 32) acc[t] = 0.0f;
        __syncthreads();

        for (int u = 0; u < NUITER; ++u) {
            // ============ phase 1: two 16-pair tiles through WMMA ============
            #pragma unroll
            for (int t2 = 0; t2 < 2; ++t2) {
                const int p = 32 * u + 16 * t2 + lh;   // pair for this lane (both halves)
                float d = 0.0f, env = 0.0f;
                if (p < NPAIR) {
                    int i  = p / 14;
                    int jj = p - 14 * i;
                    int j  = jj + (jj >= i ? 1 : 0);
                    float dx = xs[j * 3 + 0] - xs[i * 3 + 0];
                    float dy = xs[j * 3 + 1] - xs[i * 3 + 1];
                    float dz = xs[j * 3 + 2] - xs[i * 3 + 2];
                    d = __builtin_amdgcn_sqrtf(dx * dx + dy * dy + dz * dz);
                    float x = d * (1.0f / CUTOFFF);
                    if (x <= 1.0f) {
                        float x2 = x * x, x3 = x2 * x;
                        env = 1.0f - 6.0f * x3 * x2 + 15.0f * x2 * x2 - 10.0f * x3;
                    }
                }
                // A-fragments: per slot, 1 fma + 1 exp2 + 2 mul
                v16h a0, a1;
                #pragma unroll
                for (int h = 0; h < 16; ++h) {
                    {
                        float t = __builtin_fmaf(d, isg[h], nmu[h]);
                        a0[h] = (_Float16)(env * __builtin_amdgcn_exp2f(-(t * t)));
                    }
                    {
                        float t = __builtin_fmaf(d, isg[16 + h], nmu[16 + h]);
                        a1[h] = (_Float16)(env * __builtin_amdgcn_exp2f(-(t * t)));
                    }
                }
                // C initialized with bias (all D elements in this lane share N = lh)
                v8f cacc;
                #pragma unroll
                for (int r = 0; r < 8; ++r) cacc[r] = b0v;
                // EXEC is all-1s here (branches above reconverged)
                cacc = __builtin_amdgcn_wmma_f32_16x16x32_f16(
                    false, a0, false, bf0, (short)0, cacc, false, false);
                cacc = __builtin_amdgcn_wmma_f32_16x16x32_f16(
                    false, a1, false, bf1, (short)0, cacc, false, false);
                // ssp + transpose-out: D lane layout row M = r + 8*hi, col N = lh
                #pragma unroll
                for (int r = 0; r < 8; ++r) {
                    int row = 16 * t2 + r + (hi ? 8 : 0);
                    h1buf[row * 17 + lh] = ssp(cacc[r]);
                }
            }
            __syncthreads();

            // ============ phase 2: per-pair 16->4->1 MLP + backflow ============
            const int p = 32 * u + l;
            if (p < NPAIR) {
                float h1v[16];
                #pragma unroll
                for (int f = 0; f < 16; ++f) h1v[f] = h1buf[l * 17 + f];
                float h2[4];
                #pragma unroll
                for (int o = 0; o < 4; ++o) h2[o] = b1p[o];
                #pragma unroll
                for (int f = 0; f < 16; ++f) {
                    #pragma unroll
                    for (int o = 0; o < 4; ++o)
                        h2[o] = __builtin_fmaf(h1v[f], w1p[f * 4 + o], h2[o]);
                }
                float w = 0.0f;
                #pragma unroll
                for (int o = 0; o < 4; ++o) w = __builtin_fmaf(ssp(h2[o]), w2p[o], w);

                int i  = p / 14;
                int jj = p - 14 * i;
                int j  = jj + (jj >= i ? 1 : 0);
                #pragma unroll
                for (int c = 0; c < 3; ++c) {
                    float diff = xs[j * 3 + c] - xs[i * 3 + c];
                    atomicAdd(&acc[i * 3 + c], w * diff);   // ds_add_f32
                }
            }
            __syncthreads();
        }

        // xs += backflow displacement
        for (int t = l; t < NPART * 3; t += 32) xs[t] = xs[t] + acc[t];
        __syncthreads();
    }

    float* obase = out + (size_t)blk * (NPART * 3);
    for (int t = l; t < NPART * 3; t += 32) obase[t] = xs[t];
}

extern "C" void kernel_launch(void* const* d_in, const int* in_sizes, int n_in,
                              void* d_out, int out_size, void* d_ws, size_t ws_size,
                              hipStream_t stream) {
    const float* rs = (const float*)d_in[0];
    const float* W0 = (const float*)d_in[1];
    const float* b0 = (const float*)d_in[2];
    const float* W1 = (const float*)d_in[3];
    const float* b1 = (const float*)d_in[4];
    const float* W2 = (const float*)d_in[5];
    float* out = (float*)d_out;

    const int batch = in_sizes[0] / (2 * NPART * 3);   // 4096
    dim3 grid(batch * 2);                              // one wave per (batch, spin)
    dim3 block(32);
    hipLaunchKernelGGL(Backflow_70471823393186_kernel, grid, block, 0, stream,
                       rs, W0, b0, W1, b1, W2, out);
}
